// NeuralDecisionTree_37314675868533
// MI455X (gfx1250) — matrix operand order repaired and checked by reading, not compiled, not run on hardware
//
#include <hip/hip_runtime.h>
#include <hip/hip_bf16.h>
#include <stdint.h>

// ---------------------------------------------------------------------------
// NeuralDecisionTree forward for MI455X (gfx1250, wave32, WMMA).
//   out[b,o] = sum_l path[b,l] * ( relu(x[b,:] @ W1[l] + b1[l]) @ W2[l] + b2[l] )[o]
// Compute-bound (~37 GFLOP vs ~20MB traffic) -> all GEMMs via
// v_wmma_f32_16x16x32_bf16. Transposed formulation (h^T = W1^T @ x^T,
// out^T = W2^T @ h^T) keeps the batch dim in lanes so GEMM1's D feeds GEMM2's
// B with only a half-wave shuffle. W1/W2 fragments are streamed into LDS with
// the Tensor Data Mover (double buffered, s_wait_tensorcnt).
// ---------------------------------------------------------------------------

#define NUM_LEAVES 64
#define NUM_NODES  63
#define D_IN       128
#define HDIM       64
#define D_OUT      10
#define BATCH      32768

typedef __attribute__((ext_vector_type(16))) __bf16        v16bf;
typedef __attribute__((ext_vector_type(8)))  float         v8f;
typedef __attribute__((ext_vector_type(4)))  unsigned int  v4u;
typedef __attribute__((ext_vector_type(8)))  int           v8i;
typedef __attribute__((ext_vector_type(4)))  int           v4i;

// Workspace layout (bytes)
#define WR_OFF     0                            // Wr^T A-frags: 4t x 4kc x 32lane x 16 halfs bf16 = 16KB
#define W1W2_OFF   16384                        // per leaf: W1^T frags 16KB + W2^T frags 2KB
#define LEAF_BYTES 18432
#define B2P_OFF    (16384 + 64*18432)           // b2 padded 64x16 f32 = 4KB
#define PATHT_OFF  (B2P_OFF + 4096)             // pathT[64][BATCH] f32 = 8MB

// A-operand (16-bit, 16x32) slot->K mapping (matches B operands):
//   lanes0-15: slots hold K=0..15 (2/VGPR); lanes16-31: K=16..31
__device__ __forceinline__ int kOfSlot(int s, int half) {
  return s + 16 * half;
}

// ---------------------------------------------------------------------------
// Pack kernel: reformat Wr/W1/W2 (f32 row-major) into bf16 WMMA A-fragments
// and pad b2 to 16 outputs. Runs every launch (deterministic, tiny).
// ---------------------------------------------------------------------------
__global__ __launch_bounds__(256) void ndt_pack_kernel(
    const float* __restrict__ Wr, const float* __restrict__ W1,
    const float* __restrict__ W2, const float* __restrict__ b2,
    char* __restrict__ ws) {
  int idx = blockIdx.x * 256 + threadIdx.x;
  __bf16* wrp = (__bf16*)(ws + WR_OFF);
  __bf16* w12 = (__bf16*)(ws + W1W2_OFF);
  float*  b2p = (float*)(ws + B2P_OFF);
  if (idx < 8192) {                       // Wr^T frags (route GEMM A-operand)
    int s = idx & 15, lane = (idx >> 4) & 31, kc = (idx >> 9) & 3, t = (idx >> 11) & 3;
    int rowA = 16 * t + (lane & 15);      // A row = tree node
    int k = 32 * kc + kOfSlot(s, lane >> 4);
    float v = (rowA < NUM_NODES) ? Wr[rowA * D_IN + k] : 0.0f;
    wrp[idx] = (__bf16)v;
  } else if (idx < 8192 + 64 * 9216) {    // W1^T + W2^T frags per leaf
    int j = idx - 8192;
    int l = j / 9216, r = j % 9216;
    float v;
    if (r < 8192) {                       // W1^T: A row = hidden unit
      int s = r & 15, lane = (r >> 4) & 31, kc = (r >> 9) & 3, t = (r >> 11) & 3;
      int rowA = 16 * t + (lane & 15);
      int k = 32 * kc + kOfSlot(s, lane >> 4);
      v = W1[(l * D_IN + k) * HDIM + rowA];
    } else {                              // W2^T: A row = output dim (pad 10->16)
      int r2 = r - 8192;
      int s = r2 & 15, lane = (r2 >> 4) & 31, kc2 = (r2 >> 9) & 1;
      int o = lane & 15;
      int k = 32 * kc2 + kOfSlot(s, lane >> 4);
      v = (o < D_OUT) ? W2[(l * HDIM + k) * D_OUT + o] : 0.0f;
    }
    w12[j] = (__bf16)v;
  } else if (idx < 8192 + 64 * 9216 + 1024) {  // b2 padded
    int m = idx - (8192 + 64 * 9216);
    int l = m >> 4, o = m & 15;
    b2p[m] = (o < D_OUT) ? b2[l * D_OUT + o] : 0.0f;
  }
}

// ---------------------------------------------------------------------------
// Load B-operand fragments of x^T (128x16) for one 16-row batch tile.
// B-layout used here (16-bit 32x16, contiguous k, same as A packing): lanes0-15 hold K=0..15
// (s==k, 2/VGPR), lanes16-31 hold K=16..31; N = lane&15 (batch row).
// => lane reads 16 contiguous f32 from its row.
// ---------------------------------------------------------------------------
__device__ __forceinline__ void load_x_frags(const float* __restrict__ x,
                                             int row, int half, v16bf xb[4]) {
#pragma unroll
  for (int kc = 0; kc < 4; ++kc) {
    const float* xr = x + (size_t)row * D_IN + kc * 32 + 16 * half;
    v16bf v;
#pragma unroll
    for (int i = 0; i < 16; ++i) v[i] = (__bf16)xr[i];
    xb[kc] = v;
  }
}

// ---------------------------------------------------------------------------
// Route+path kernel: route^T = sigmoid(Wr^T @ x^T + br); expand to 64 leaf
// path products; store pathT[leaf][batch] (coalesced for the main kernel).
// ---------------------------------------------------------------------------
__global__ __launch_bounds__(256) void ndt_route_kernel(
    const float* __restrict__ x, const float* __restrict__ br,
    char* __restrict__ ws) {
  const __bf16* wrp = (const __bf16*)(ws + WR_OFF);
  float* pathT = (float*)(ws + PATHT_OFF);
  int lane = threadIdx.x & 31;
  int wave = threadIdx.x >> 5;
  int half = lane >> 4;
  int tile = blockIdx.x * 8 + wave;
  int row  = tile * 16 + (lane & 15);

  v16bf xb[4];
  load_x_frags(x, row, half, xb);

  float sig[32];
#pragma unroll
  for (int t = 0; t < 4; ++t) {
    v8f a = {};
#pragma unroll
    for (int kc = 0; kc < 4; ++kc) {
      v16bf A = *(const v16bf*)(wrp + ((t * 4 + kc) * 32 + lane) * 16);
      a = __builtin_amdgcn_wmma_f32_16x16x32_bf16(false, A, false, xb[kc],
                                                  (short)0, a, false, false);
    }
#pragma unroll
    for (int r = 0; r < 8; ++r) {
      int node = 16 * t + 8 * half + r;     // D: VGPR r -> node, lane -> batch
      float bias = (node < NUM_NODES) ? br[node] : 0.0f;
      float z = a[r] + bias;
      sig[t * 8 + r] = 1.0f / (1.0f + __expf(-z));
    }
  }
  // exchange half-waves so each lane owns all 63 node probabilities of its row
  float route[64];
#pragma unroll
  for (int t = 0; t < 4; ++t) {
#pragma unroll
    for (int r = 0; r < 8; ++r) {
      float own = sig[t * 8 + r];
      float oth = __shfl_xor(own, 16, 32);
      route[16 * t + r]     = half ? oth : own;
      route[16 * t + 8 + r] = half ? own : oth;
    }
  }
  if (lane < 16) {
#pragma unroll
    for (int L = 0; L < 64; ++L) {
      float p = 1.0f;
#pragma unroll
      for (int v = 0; v < 6; ++v) {
        int node = (1 << v) - 1 + (L >> (6 - v));   // compile-time per L
        int bit  = (L >> (5 - v)) & 1;              // 0 = left (d), 1 = right
        float d = route[node];
        p *= bit ? (1.0f - d) : d;
      }
      pathT[(size_t)L * BATCH + row] = p;
    }
  }
}

// ---------------------------------------------------------------------------
// TDM: stream one leaf's packed W1/W2 (18KB, 4608 dwords) into LDS.
// Dynamic shared memory assumed at LDS offset 0 (no static LDS in kernel).
// ---------------------------------------------------------------------------
#if defined(__has_builtin)
#if __has_builtin(__builtin_amdgcn_tensor_load_to_lds) && \
    __has_builtin(__builtin_amdgcn_s_wait_tensorcnt)
#define HAVE_TDM 1
#endif
#endif
#ifndef HAVE_TDM
#define HAVE_TDM 0
#endif

#if HAVE_TDM
__device__ __forceinline__ void tdm_load_leaf(const char* gsrc, unsigned lds_off) {
  uint64_t ga = (uint64_t)(uintptr_t)gsrc;
  v4u g0;
  g0[0] = 1u;                                           // count=1 (valid D#)
  g0[1] = lds_off;                                      // lds_addr
  g0[2] = (unsigned)(ga & 0xffffffffu);                 // global_addr[31:0]
  g0[3] = (unsigned)((ga >> 32) & 0x01ffffffu) | (2u << 30);  // addr[56:32] | type=2
  v8i g1;
  g1[0] = (2 << 16);                                    // data_size=4B, wg_mask=0
  g1[1] = (int)(4608u << 16);                           // tensor_dim0[15:0]
  g1[2] = (int)(1u << 16);                              // tensor_dim0 hi=0 | tensor_dim1=1
  g1[3] = (int)(4608u << 16);                           // tile_dim0=4608
  g1[4] = 1;                                            // tile_dim1=1, tile_dim2=0
  g1[5] = 4608;                                         // tensor_dim0_stride
  g1[6] = 0;
  g1[7] = 0;
  v4i z4 = {0, 0, 0, 0};
#if __clang_major__ >= 23
  v8i z8 = {0, 0, 0, 0, 0, 0, 0, 0};
  __builtin_amdgcn_tensor_load_to_lds(g0, g1, z4, z4, z8, 0);
#else
  __builtin_amdgcn_tensor_load_to_lds(g0, g1, z4, z4, 0);
#endif
}
#endif

// ---------------------------------------------------------------------------
// Main kernel: per wave one 16-row batch tile. For each leaf:
//   h^T frags (4x v8f) = sum_kc WMMA(W1^T frag, x^T frag)
//   h = relu(h + b1) * path ; repack (bf16) into GEMM2 B operand with a
//   half-wave shuffle; out^T += WMMA(W2^T frag, h^T) ; += path*b2.
// ---------------------------------------------------------------------------
__global__ __launch_bounds__(256) void ndt_main_kernel(
    const float* __restrict__ x, const float* __restrict__ b1,
    char* __restrict__ ws, float* __restrict__ out) {
  extern __shared__ char smem[];   // 2 x LEAF_BYTES double buffer (TDM path)
  const float* pathT = (const float*)(ws + PATHT_OFF);
  const float* b2pad = (const float*)(ws + B2P_OFF);
  const char*  w12   = ws + W1W2_OFF;
  int lane = threadIdx.x & 31;
  int wave = threadIdx.x >> 5;
  int half = lane >> 4;
  int tile = blockIdx.x * 8 + wave;
  int row  = tile * 16 + (lane & 15);

  v16bf xb[4];
  load_x_frags(x, row, half, xb);
  v8f oacc = {};

#if HAVE_TDM
  if (wave == 0) tdm_load_leaf(w12, 0);   // prologue: leaf 0 -> buffer 0
#endif

#pragma unroll 1
  for (int l = 0; l < NUM_LEAVES; ++l) {
#if HAVE_TDM
    if (wave == 0) {
      if (l + 1 < NUM_LEAVES) {
        tdm_load_leaf(w12 + (size_t)(l + 1) * LEAF_BYTES,
                      (unsigned)(((l + 1) & 1) * LEAF_BYTES));
        __builtin_amdgcn_s_wait_tensorcnt(1);   // leaf l complete, l+1 in flight
      } else {
        __builtin_amdgcn_s_wait_tensorcnt(0);
      }
    }
    __syncthreads();                            // buffer (l&1) ready for all waves
    const char* wbuf = smem + (l & 1) * LEAF_BYTES;
#else
    __syncthreads();                            // previous compute done
    {
      const float4* src = (const float4*)(w12 + (size_t)l * LEAF_BYTES);
      float4* dst = (float4*)smem;
      for (int i = threadIdx.x; i < LEAF_BYTES / 16; i += 256) dst[i] = src[i];
    }
    __syncthreads();
    const char* wbuf = smem;
#endif

    // ---- GEMM1': h^T(64x16) = W1^T @ x^T, 4 M-tiles x 4 K-chunks ----
    v8f hacc[4];
#pragma unroll
    for (int t = 0; t < 4; ++t) {
      v8f a = {};
#pragma unroll
      for (int kc = 0; kc < 4; ++kc) {
        v16bf A = *(const v16bf*)(wbuf + (t * 4 + kc) * 1024 + lane * 32);
        a = __builtin_amdgcn_wmma_f32_16x16x32_bf16(false, A, false, xb[kc],
                                                    (short)0, a, false, false);
      }
      hacc[t] = a;
    }

    // ---- bias + relu + path scale, pack bf16 pairs ----
    float pathv = pathT[(size_t)l * BATCH + row];
    unsigned int hp[4][4];
#pragma unroll
    for (int t = 0; t < 4; ++t) {
      const float* b1p = b1 + l * HDIM + t * 16 + 8 * half;  // hidden = 16t+8*half+r
#pragma unroll
      for (int d = 0; d < 4; ++d) {
        float f0 = fmaxf(hacc[t][2 * d]     + b1p[2 * d],     0.0f) * pathv;
        float f1 = fmaxf(hacc[t][2 * d + 1] + b1p[2 * d + 1], 0.0f) * pathv;
        __bf16 h0 = (__bf16)f0, h1 = (__bf16)f1;
        hp[t][d] = (unsigned)__builtin_bit_cast(unsigned short, h0) |
                   ((unsigned)__builtin_bit_cast(unsigned short, h1) << 16);
      }
    }

    // ---- GEMM2: out^T(16x16) += W2^T @ h^T, 2 K-chunks of 32 ----
    // B needs lanes0-15: k = 32c+0..15, lanes16-31: k = 32c+16..31.
    // Lane owns hidden {16t+8*half+r}; the other 8 values per chunk come from
    // the partner half-wave via shfl_xor(16).
#pragma unroll
    for (int c = 0; c < 2; ++c) {
      union { unsigned int u[8]; v16bf v; } B;
#pragma unroll
      for (int d = 0; d < 4; ++d) {
        unsigned lo  = hp[2 * c][d];                       // hidden 32c+8*half+2d..
        unsigned hi2 = hp[2 * c + 1][d];                   // hidden 32c+16+8*half+2d..
        unsigned xlo = (unsigned)__shfl_xor((int)lo, 16, 32);
        unsigned xhi = (unsigned)__shfl_xor((int)hi2, 16, 32);
        B.u[d]     = half ? xhi : lo;    // k = 32c+(16*half)+2d..
        B.u[d + 4] = half ? hi2 : xlo;   // k = 32c+(16*half)+8+2d..
      }
      v16bf A2 = *(const v16bf*)(wbuf + 16384 + c * 1024 + lane * 32);
      oacc = __builtin_amdgcn_wmma_f32_16x16x32_bf16(false, A2, false, B.v,
                                                     (short)0, oacc, false, false);
    }

    // ---- += path * b2 (out dim o = r + 8*half) ----
    const float* b2p = b2pad + l * 16 + 8 * half;
#pragma unroll
    for (int r = 0; r < 8; ++r) oacc[r] += pathv * b2p[r];

#if HAVE_TDM
    __syncthreads();   // buffer (l&1) free before wave0 reissues into it
#endif
  }

  // out^T D-layout: lane -> batch row, VGPR r -> output o = r + 8*half
#pragma unroll
  for (int r = 0; r < 8; ++r) {
    int o = r + 8 * half;
    if (o < D_OUT) out[(size_t)row * D_OUT + o] = oacc[r];
  }
}

// ---------------------------------------------------------------------------
extern "C" void kernel_launch(void* const* d_in, const int* in_sizes, int n_in,
                              void* d_out, int out_size, void* d_ws, size_t ws_size,
                              hipStream_t stream) {
  const float* x  = (const float*)d_in[0];
  const float* Wr = (const float*)d_in[1];
  const float* br = (const float*)d_in[2];
  const float* W1 = (const float*)d_in[3];
  const float* b1 = (const float*)d_in[4];
  const float* W2 = (const float*)d_in[5];
  const float* b2 = (const float*)d_in[6];
  float* out = (float*)d_out;
  char* ws = (char*)d_ws;

  // 1) reformat weights into WMMA fragment layout (bf16) + padded b2
  int pack_elems = 8192 + 64 * 9216 + 1024;
  ndt_pack_kernel<<<(pack_elems + 255) / 256, 256, 0, stream>>>(Wr, W1, W2, b2, ws);

  // 2) routing probabilities -> leaf path weights, transposed
  ndt_route_kernel<<<BATCH / 128, 256, 0, stream>>>(x, br, ws);

  // 3) main fused leaf GEMMs (8 waves/block, 128 batch rows/block)
  ndt_main_kernel<<<BATCH / 128, 256, 2 * LEAF_BYTES, stream>>>(x, b1, ws, out);
}
